// MultiHeadedAttention_28372553957625
// MI455X (gfx1250) — matrix operand order repaired and checked by reading, not compiled, not run on hardware
//
#include <hip/hip_runtime.h>

// ---- problem constants (match reference) ----
#define BS 1
#define SS 4096
#define DD 512
#define HH 8
#define PD 64   // DD / HH

typedef __attribute__((ext_vector_type(16))) _Float16 v16h;
typedef __attribute__((ext_vector_type(8)))  float    v8f;

// ---------------- WMMA fragment helpers (CDNA5 wave32 layouts) ----------------
// A (16x32 f16): lane m = L%16 holds row m; khalf = L/16 adds +16 to K.
// element e -> K = e + 16*khalf (matches B's contiguous order)
// 16 contiguous K values -> one 32-byte contiguous load.
__device__ __forceinline__ v16h load_a_f16(const _Float16* A, int lda, int lane, int kofs) {
    const int m  = lane & 15;
    const int kh = (lane >> 4) & 1;
    const _Float16* p = A + (size_t)m * lda + kofs + kh * 16;
    v16h a;
#pragma unroll
    for (int e = 0; e < 16; ++e) a[e] = p[e];
    return a;
}

// B (32x16 f16) from a [N][K] (transposed) layout: lane n = L%16 holds column n,
// 16 contiguous K values starting at 16*khalf  -> one 32-byte contiguous load.
__device__ __forceinline__ v16h load_b_f16(const _Float16* Bt, int ldb, int lane) {
    const int n  = lane & 15;
    const int kh = (lane >> 4) & 1;
    const _Float16* p = Bt + (size_t)n * ldb + kh * 16;
    v16h b;
#pragma unroll
    for (int e = 0; e < 16; ++e) b[e] = p[e];
    return b;
}

#define WMMA_F16(a, b, c) \
    __builtin_amdgcn_wmma_f32_16x16x32_f16(false, (a), false, (b), (short)0, (c), false, false)

// ---------------- kernel 0: pack W (fp32 [K][N]) -> Wt (f16 [N][K]) ----------------
__global__ void pack_wt_kernel(const float* __restrict__ W, _Float16* __restrict__ Wt) {
    int idx = blockIdx.x * blockDim.x + threadIdx.x;
    if (idx < DD * DD) {
        int n = idx / DD;
        int k = idx - n * DD;
        Wt[(size_t)n * DD + k] = (_Float16)W[(size_t)k * DD + n];
    }
}

// ---------------- kernel 1: projection GEMM  Out = X(f32) @ W + b  (f16 out) ------
// grid = (S/16, D/128), block = 256 (8 waves). Wave w covers cols cg*128 + w*16.
__global__ void proj_gemm_kernel(const float* __restrict__ X,
                                 const _Float16* __restrict__ Wt,  // [N][K] f16
                                 const float* __restrict__ bias,
                                 _Float16* __restrict__ Out,
                                 int transposed_out) {
    extern __shared__ char smem_raw[];
    _Float16* As = (_Float16*)smem_raw;          // 16 x 512 f16 = 16 KB

    const int r0  = blockIdx.x * 16;
    const int cg  = blockIdx.y;
    const int tid = threadIdx.x;

    // stage A tile, converting fp32 -> f16
    for (int i = tid; i < 16 * DD; i += 256) {
        int r = i >> 9;          // /512
        int c = i & (DD - 1);
        As[i] = (_Float16)X[(size_t)(r0 + r) * DD + c];
    }
    __syncthreads();

    const int wave = tid >> 5;
    const int lane = tid & 31;
    const int col0 = cg * 128 + wave * 16;

    v8f acc = {};
#pragma unroll
    for (int t = 0; t < DD / 32; ++t) {
        v16h a = load_a_f16(As, DD, lane, t * 32);
        v16h b = load_b_f16(Wt + (size_t)col0 * DD + t * 32, DD, lane);
        acc = WMMA_F16(a, b, acc);
    }

    const int n  = lane & 15;
    const int kh = lane >> 4;
    const float bv = bias[col0 + n];
#pragma unroll
    for (int r = 0; r < 8; ++r) {
        const float val = acc[r] + bv;
        const int row = r0 + r + 8 * kh;
        if (transposed_out)
            Out[(size_t)(col0 + n) * SS + row] = (_Float16)val;   // Vt[D][S]
        else
            Out[(size_t)row * DD + col0 + n] = (_Float16)val;     // [S][D]
    }
}

// ---------------- kernel 2: attention core, one block per (16-row tile, head) ------
// LDS: 16 x 4096 f16 score strip (128 KB) + 4 KB reduction scratch.
// 128 KB/block -> two blocks co-resident per 320 KB WGP.
__global__ void attn_kernel(const _Float16* __restrict__ Qb,   // [S][D]
                            const _Float16* __restrict__ Kb,   // [S][D]
                            const _Float16* __restrict__ Vt,   // [D][S]
                            float* __restrict__ Wout,          // [H][S][S]
                            _Float16* __restrict__ merged) {   // [S][D]
    extern __shared__ char smem_raw[];
    _Float16* sc  = (_Float16*)smem_raw;                       // 16*4096 f16
    float*    red = (float*)(smem_raw + (size_t)16 * SS * sizeof(_Float16));

    const int r0   = blockIdx.x * 16;
    const int h    = blockIdx.y;
    const int tid  = threadIdx.x;
    const int wave = tid >> 5;
    const int lane = tid & 31;

    // ---- phase 1: scores = (Q K^T) * 1/sqrt(PD) -> f16 LDS strip ----
    const _Float16* Qbase = Qb + (size_t)r0 * DD + h * PD;
    v16h aq0 = load_a_f16(Qbase, DD, lane, 0);
    v16h aq1 = load_a_f16(Qbase, DD, lane, 32);

    for (int tile = 0; tile < SS / (16 * 8); ++tile) {
        const int n0 = (tile * 8 + wave) * 16;
        if (n0 + 256 < SS)   // prefetch K rows 16 tiles ahead (global_prefetch_b8)
            __builtin_prefetch(Kb + (size_t)(n0 + 256) * DD + h * PD, 0, 0);

        v16h bk0 = load_b_f16(Kb + (size_t)n0 * DD + h * PD,      DD, lane);
        v16h bk1 = load_b_f16(Kb + (size_t)n0 * DD + h * PD + 32, DD, lane);
        v8f c = {};
        c = WMMA_F16(aq0, bk0, c);
        c = WMMA_F16(aq1, bk1, c);

        const int n  = lane & 15;
        const int kh = lane >> 4;
#pragma unroll
        for (int r = 0; r < 8; ++r)
            sc[(size_t)(r + 8 * kh) * SS + n0 + n] = (_Float16)(c[r] * 0.125f);
    }
    __syncthreads();

    // ---- phase 2: row softmax; fp32 weights to global (float4 stores),
    //      normalized f16 weights kept in place in LDS for phase 3 ----
    {
        const int row = tid >> 4;       // 0..15
        const int j   = tid & 15;       // owns 4 contiguous cols per 64-stride step
        _Float16* srow = sc + (size_t)row * SS;

        float mx = -3.402823466e+38f;
        for (int c = j * 4; c < SS; c += 64) {
#pragma unroll
            for (int u = 0; u < 4; ++u) mx = fmaxf(mx, (float)srow[c + u]);
        }
#pragma unroll
        for (int off = 1; off < 16; off <<= 1) mx = fmaxf(mx, __shfl_xor(mx, off, 32));

        float sum = 0.0f;
        for (int c = j * 4; c < SS; c += 64) {
#pragma unroll
            for (int u = 0; u < 4; ++u) {
                float e = __expf((float)srow[c + u] - mx);
                srow[c + u] = (_Float16)e;
                sum += e;
            }
        }
#pragma unroll
        for (int off = 1; off < 16; off <<= 1) sum += __shfl_xor(sum, off, 32);
        const float inv = 1.0f / sum;

        float* wrow = Wout + ((size_t)h * SS + (r0 + row)) * SS;
        for (int c = j * 4; c < SS; c += 64) {
            float4 wv;
            float w0 = (float)srow[c + 0] * inv;
            float w1 = (float)srow[c + 1] * inv;
            float w2 = (float)srow[c + 2] * inv;
            float w3 = (float)srow[c + 3] * inv;
            srow[c + 0] = (_Float16)w0;
            srow[c + 1] = (_Float16)w1;
            srow[c + 2] = (_Float16)w2;
            srow[c + 3] = (_Float16)w3;
            wv.x = w0; wv.y = w1; wv.z = w2; wv.w = w3;
            *(float4*)(wrow + c) = wv;       // 16B global store, 256B/row-group
        }
    }
    __syncthreads();

    // ---- phase 3: attn = weights @ V, all 8 waves (K split in halves) ----
    {
        const int nt  = wave & 3;            // PD 16-col slice
        const int nc0 = nt * 16;
        const int ks  = (wave >> 2) * (SS / 2);
        const _Float16* Vbase = Vt + (size_t)(h * PD + nc0) * SS;

        v8f acc = {};
        for (int kt = 0; kt < SS / 64; ++kt) {       // 64 iters per wave
            const int k0 = ks + kt * 32;
            v16h aw = load_a_f16(sc, SS, lane, k0);  // pure ds_load_b128, no cvt
            v16h bv = load_b_f16(Vbase + k0, SS, lane);
            acc = WMMA_F16(aw, bv, acc);
        }

        if (wave >= 4) {                 // upper-half partials -> LDS scratch
            float* dst = red + ((size_t)(wave - 4) * 32 + lane) * 8;
#pragma unroll
            for (int r = 0; r < 8; ++r) dst[r] = acc[r];
        }
        __syncthreads();
        if (wave < 4) {
            const float* src = red + ((size_t)wave * 32 + lane) * 8;
            const int n  = lane & 15;
            const int kh = lane >> 4;
#pragma unroll
            for (int r = 0; r < 8; ++r) {
                float v = acc[r] + src[r];
                merged[(size_t)(r0 + r + 8 * kh) * DD + h * PD + nc0 + n] = (_Float16)v;
            }
        }
    }
}

// ---------------- kernel 3: output GEMM  out = merged(f16) @ Wo + bo (fp32) -------
__global__ void out_gemm_kernel(const _Float16* __restrict__ Xb,   // [S][D] f16
                                const _Float16* __restrict__ Wt,   // [N][K] f16
                                const float* __restrict__ bias,
                                float* __restrict__ Out) {
    extern __shared__ char smem_raw[];
    _Float16* As = (_Float16*)smem_raw;

    const int r0  = blockIdx.x * 16;
    const int cg  = blockIdx.y;
    const int tid = threadIdx.x;

    for (int i = tid; i < 16 * DD; i += 256) {
        int r = i >> 9;
        int c = i & (DD - 1);
        As[i] = Xb[(size_t)(r0 + r) * DD + c];
    }
    __syncthreads();

    const int wave = tid >> 5;
    const int lane = tid & 31;
    const int col0 = cg * 128 + wave * 16;

    v8f acc = {};
#pragma unroll
    for (int t = 0; t < DD / 32; ++t) {
        v16h a = load_a_f16(As, DD, lane, t * 32);
        v16h b = load_b_f16(Wt + (size_t)col0 * DD + t * 32, DD, lane);
        acc = WMMA_F16(a, b, acc);
    }

    const int n  = lane & 15;
    const int kh = lane >> 4;
    const float bv = bias[col0 + n];
#pragma unroll
    for (int r = 0; r < 8; ++r)
        Out[(size_t)(r0 + r + 8 * kh) * DD + col0 + n] = acc[r] + bv;
}

// ---------------- host-side launch sequence ----------------
extern "C" void kernel_launch(void* const* d_in, const int* in_sizes, int n_in,
                              void* d_out, int out_size, void* d_ws, size_t ws_size,
                              hipStream_t stream) {
    const float* x1 = (const float*)d_in[0];
    const float* x2 = (const float*)d_in[1];
    const float* x3 = (const float*)d_in[2];
    const float* Wq = (const float*)d_in[3];
    const float* bq = (const float*)d_in[4];
    const float* Wk = (const float*)d_in[5];
    const float* bk = (const float*)d_in[6];
    const float* Wv = (const float*)d_in[7];
    const float* bv = (const float*)d_in[8];
    const float* Wo = (const float*)d_in[9];
    const float* bo = (const float*)d_in[10];

    float* out_proj = (float*)d_out;                       // [S][D]
    float* out_wts  = (float*)d_out + (size_t)SS * DD;     // [H][S][S]

    // workspace layout (f16 elements)
    _Float16* ws = (_Float16*)d_ws;
    const size_t WSZ = (size_t)DD * DD;   // 262144
    const size_t MSZ = (size_t)SS * DD;   // 2097152
    _Float16* Wtq = ws;
    _Float16* Wtk = Wtq + WSZ;
    _Float16* Wtv = Wtk + WSZ;
    _Float16* Wto = Wtv + WSZ;
    _Float16* Qb  = Wto + WSZ;
    _Float16* Kb  = Qb + MSZ;
    _Float16* Vt  = Kb + MSZ;   // transposed V: [D][S]
    _Float16* Mg  = Vt + MSZ;   // merged attn output [S][D]

    // 0) pack weights to f16 [N][K]
    const int pw_grid = (DD * DD + 255) / 256;
    pack_wt_kernel<<<pw_grid, 256, 0, stream>>>(Wq, Wtq);
    pack_wt_kernel<<<pw_grid, 256, 0, stream>>>(Wk, Wtk);
    pack_wt_kernel<<<pw_grid, 256, 0, stream>>>(Wv, Wtv);
    pack_wt_kernel<<<pw_grid, 256, 0, stream>>>(Wo, Wto);

    // 1) Q/K/V projections (V stored transposed)
    dim3 pg(SS / 16, DD / 128);
    const size_t proj_smem = 16 * DD * sizeof(_Float16);   // 16 KB
    proj_gemm_kernel<<<pg, 256, proj_smem, stream>>>(x1, Wtq, bq, Qb, 0);
    proj_gemm_kernel<<<pg, 256, proj_smem, stream>>>(x2, Wtk, bk, Kb, 0);
    proj_gemm_kernel<<<pg, 256, proj_smem, stream>>>(x3, Wtv, bv, Vt, 1);

    // 2) attention (f16 score strip: 128 KB LDS + 4 KB scratch -> 2 blocks/WGP)
    dim3 ag(SS / 16, HH);
    const size_t attn_smem = (size_t)16 * SS * sizeof(_Float16) + 4 * 32 * 8 * sizeof(float);
    attn_kernel<<<ag, 256, attn_smem, stream>>>(Qb, Kb, Vt, out_wts, Mg);

    // 3) output projection
    out_gemm_kernel<<<pg, 256, proj_smem, stream>>>(Mg, Wto, bo, out_proj);
}
